// GNN_52080773431963
// MI455X (gfx1250) — matrix operand order repaired and checked by reading, not compiled, not run on hardware
//
#include <hip/hip_runtime.h>
#include <hip/hip_bf16.h>
#include <math.h>

// ---------------------------------------------------------------------------
// Types for CDNA5 WMMA
// ---------------------------------------------------------------------------
typedef __attribute__((ext_vector_type(16))) __bf16 v16bf;
typedef __attribute__((ext_vector_type(8)))  float  v8f;
typedef __attribute__((ext_vector_type(4)))  unsigned int u32x4;
typedef __attribute__((ext_vector_type(4)))  int          i32x4;

#define GNN_AS1 __attribute__((address_space(1)))
#define GNN_AS3 __attribute__((address_space(3)))

union Frag16 {
    u32x4 q[2];
    v16bf v;
};

// Optional gfx1250 async global->LDS path (probe via __has_builtin; the
// contiguous-copy fallback compiles identically if the builtin is absent).
#if defined(__has_builtin)
#  if __has_builtin(__builtin_amdgcn_global_load_async_to_lds_b128)
#    define GNN_ASYNC_LDS 1
#  endif
#  if __has_builtin(__builtin_amdgcn_s_wait_asynccnt)
#    define GNN_WAIT_ASYNC_BUILTIN 1
#  endif
#endif

__device__ __forceinline__ void gnn_wait_async() {
#ifdef GNN_ASYNC_LDS
#  ifdef GNN_WAIT_ASYNC_BUILTIN
    __builtin_amdgcn_s_wait_asynccnt(0);
#  else
    asm volatile("s_wait_asynccnt 0x0" ::: "memory");
#  endif
#endif
}

__device__ __forceinline__ unsigned short f32_bf16(float f) {
    union { float f; unsigned u; } c;
    c.f = f;
    unsigned r = c.u + 0x7FFFu + ((c.u >> 16) & 1u);   // round-to-nearest-even
    return (unsigned short)(r >> 16);
}

// ---------------------------------------------------------------------------
// Utility: zero a float buffer
// ---------------------------------------------------------------------------
__global__ void zero_f32(float* __restrict__ p, long long n) {
    long long i = (long long)blockIdx.x * blockDim.x + threadIdx.x;
    if (i < n) p[i] = 0.0f;
}

// ---------------------------------------------------------------------------
// SpMM: y[r,:] += val * x[c,:]   (one wave per edge, float4 per lane)
// y must be pre-zeroed. Fdim is 128 or 256.
// ---------------------------------------------------------------------------
__global__ __launch_bounds__(256) void spmm_atomic(
    const int* __restrict__ rows, const int* __restrict__ cols,
    const float* __restrict__ vals, const float* __restrict__ x,
    float* __restrict__ y, int E, int Fdim)
{
    int wave = (blockIdx.x * blockDim.x + threadIdx.x) >> 5;
    int lane = threadIdx.x & 31;
    if (wave >= E) return;
    int r = rows[wave];
    int c = cols[wave];
    float v = vals[wave];
    const float* xr = x + (long long)c * Fdim;
    float*       yr = y + (long long)r * Fdim;
    for (int f0 = lane * 4; f0 < Fdim; f0 += 128) {
        float4 xv = *(const float4*)(xr + f0);
        atomicAdd(yr + f0 + 0, v * xv.x);
        atomicAdd(yr + f0 + 1, v * xv.y);
        atomicAdd(yr + f0 + 2, v * xv.z);
        atomicAdd(yr + f0 + 3, v * xv.w);
    }
}

// ---------------------------------------------------------------------------
// Weight prep: W6 is (6, Kin, 128) f32. Build bf16 stack of 4*Kin rows x 128
// cols, stored PRE-TRANSPOSED in 32-row blocks:
//   element (stacked row r, col c) -> Wstack[(r>>5)*4096 + c*32 + (r&31)]
// Stacked rows:
//   [0,Kin)      : W6[0]+W6[1]+W6[2]
//   [Kin,2Kin)   : W6[3] ;  [2Kin,3Kin): W6[4] ; [3Kin,4Kin): W6[5]
// ---------------------------------------------------------------------------
__global__ void wprep(const float* __restrict__ W6,
                      unsigned short* __restrict__ Wstack, int Kin)
{
    int idx = blockIdx.x * blockDim.x + threadIdx.x;
    int total = 4 * Kin * 128;
    if (idx >= total) return;
    int r = idx >> 7;          // stacked row
    int c = idx & 127;
    int s = r / Kin;
    int rl = r - s * Kin;
    float w;
    if (s == 0) {
        w = W6[(0 * Kin + rl) * 128 + c]
          + W6[(1 * Kin + rl) * 128 + c]
          + W6[(2 * Kin + rl) * 128 + c];
    } else {
        w = W6[((s + 2) * Kin + rl) * 128 + c];
    }
    Wstack[((long long)(r >> 5) << 12) + c * 32 + (r & 31)] = f32_bf16(w);
}

// ---------------------------------------------------------------------------
// WMMA GEMM:  out(N x 128) = [S0 | S1 | S2 | S3] (N x 4*Kin) @ W (4*Kin x 128)
// S* are f32 (converted to bf16 while staging); W is pre-packed/transposed
// bf16 (see wprep). Block: 256 threads = 8 waves, 128x128 output tile;
// wave w owns rows [w*16, w*16+16) x all 8 column tiles. LDS double-buffered;
// B staged via async global->LDS when available. B fragments software-
// pipelined (ba/bb) so DS loads overlap the WMMA pipe with partial waits.
// ---------------------------------------------------------------------------
__global__ __launch_bounds__(256) void gemm_wmma_bf16(
    const float* __restrict__ S0, const float* __restrict__ S1,
    const float* __restrict__ S2, const float* __restrict__ S3,
    const unsigned short* __restrict__ W, float* __restrict__ out,
    int N, int Kin, int kshift, int kmask)
{
    __shared__ u32x4 AldsV[2][512];   // 2 x (128 rows x 32 k) bf16
    __shared__ u32x4 BldsV[2][512];   // 2 x [col][k] transposed blocks

    const int tid  = threadIdx.x;
    const int lane = tid & 31;
    const int wave = tid >> 5;
    const int half = (lane >> 4) & 1;           // lane group 0-15 / 16-31
    const int l16  = lane & 15;
    const int rowBase = blockIdx.x * 128;
    const int nsteps = Kin >> 3;                // (4*Kin)/32

    // per-thread staging constants (nothing recomputed in the loop)
    const int arow0 = tid >> 4;                 // 0..15 (A row modulo 16-strip)
    const int kp2   = (tid & 15) * 2;           // bf16 pair within 32-k tile

    v8f acc[8];
#pragma unroll
    for (int t = 0; t < 8; ++t)
#pragma unroll
        for (int j = 0; j < 8; ++j) acc[t][j] = 0.0f;

#ifdef GNN_ASYNC_LDS
#define GNN_COPY_B(BUF, STEP)                                                  \
    do {                                                                       \
        u32x4* wp_ = (u32x4*)(W + ((long long)(STEP) << 12));                  \
        u32x4* Bu_ = BldsV[(BUF)];                                             \
        __builtin_amdgcn_global_load_async_to_lds_b128(                        \
            (GNN_AS1 i32x4*)(wp_ + tid),                                       \
            (GNN_AS3 i32x4*)(Bu_ + tid), 0, 0);                                \
        __builtin_amdgcn_global_load_async_to_lds_b128(                        \
            (GNN_AS1 i32x4*)(wp_ + tid + 256),                                 \
            (GNN_AS3 i32x4*)(Bu_ + tid + 256), 0, 0);                          \
    } while (0)
#else
#define GNN_COPY_B(BUF, STEP)                                                  \
    do {                                                                       \
        const u32x4* wp_ = (const u32x4*)(W + ((long long)(STEP) << 12));      \
        u32x4* Bu_ = BldsV[(BUF)];                                             \
        Bu_[tid]       = wp_[tid];                                             \
        Bu_[tid + 256] = wp_[tid + 256];                                       \
    } while (0)
#endif

#define GNN_STAGE(BUF, STEP)                                                   \
    do {                                                                       \
        const int s_ = (STEP) >> kshift;                                       \
        const float* src_ =                                                    \
            (s_ == 0) ? S0 : (s_ == 1) ? S1 : (s_ == 2) ? S2 : S3;             \
        const long long kb_ = (long long)((((STEP) & kmask) << 5) + kp2);      \
        unsigned int* Au_ = (unsigned int*)AldsV[(BUF)];                       \
        _Pragma("unroll")                                                      \
        for (int j_ = 0; j_ < 8; ++j_) {                                       \
            int gr_ = rowBase + arow0 + j_ * 16;                               \
            float2 v2_ = make_float2(0.0f, 0.0f);                              \
            if (gr_ < N) v2_ = *(const float2*)&src_[(long long)gr_ * Kin + kb_]; \
            Au_[tid + j_ * 256] =                                              \
                (unsigned)f32_bf16(v2_.x) | ((unsigned)f32_bf16(v2_.y) << 16); \
        }                                                                      \
        GNN_COPY_B(BUF, STEP);                                                 \
    } while (0)

#define GNN_LOADB(FR, CT)                                                      \
    do {                                                                       \
        (FR).q[0] = *(const u32x4*)(bbase + (CT) * 512);                       \
        (FR).q[1] = *(const u32x4*)(bbase + (CT) * 512 + 16);                  \
    } while (0)

#define GNN_MMA(CT, BF)                                                        \
    acc[CT] = __builtin_amdgcn_wmma_f32_16x16x32_bf16(                         \
        false, af.v, false, (BF).v, (short)0, acc[CT], false, false)

    GNN_STAGE(0, 0);

    for (int step = 0; step < nsteps; ++step) {
        gnn_wait_async();                    // async B copy (if used) landed
        __syncthreads();                     // staging of buffer p complete
        const int p = step & 1;
        const int nxt = (step + 1 < nsteps) ? step + 1 : step;  // clamped
        GNN_STAGE(p ^ 1, nxt);

        const unsigned short* Alds = (const unsigned short*)AldsV[p];
        const unsigned short* Blds = (const unsigned short*)BldsV[p];

        // A fragment (16x32 bf16): elems 0..7 <-> k = half*8 + 0..7,
        //                          elems 8..15 <-> k = 16 + half*8 + 0..7
        const int arow = wave * 16 + l16;
        Frag16 af;
        af.q[0] = *(const u32x4*)&Alds[arow * 32 + half * 8];
        af.q[1] = *(const u32x4*)&Alds[arow * 32 + 16 + half * 8];

        // B fragment base for this lane; column tile ct is +ct*512 ushorts
        const unsigned short* bbase = &Blds[l16 * 32 + half * 8];

        Frag16 ba, bb;
        GNN_LOADB(ba, 0);
        GNN_LOADB(bb, 1); GNN_MMA(0, ba);
        GNN_LOADB(ba, 2); GNN_MMA(1, bb);
        GNN_LOADB(bb, 3); GNN_MMA(2, ba);
        GNN_LOADB(ba, 4); GNN_MMA(3, bb);
        GNN_LOADB(bb, 5); GNN_MMA(4, ba);
        GNN_LOADB(ba, 6); GNN_MMA(5, bb);
        GNN_LOADB(bb, 7); GNN_MMA(6, ba);
        GNN_MMA(7, bb);
    }

    // Epilogue: C/D layout — VGPR j: M = j + half*8, N = l16 (+16*ct)
#pragma unroll
    for (int ct = 0; ct < 8; ++ct) {
#pragma unroll
        for (int j = 0; j < 8; ++j) {
            int gr = rowBase + wave * 16 + j + half * 8;
            if (gr < N) out[(long long)gr * 128 + ct * 16 + l16] = acc[ct][j];
        }
    }
}

// ---------------------------------------------------------------------------
// BN stats: per-feature sum / sumsq over rows (optionally after relu)
// blockDim = 128 (one thread per feature), each block covers 256 rows
// ---------------------------------------------------------------------------
__global__ __launch_bounds__(128) void bn_stats(
    const float* __restrict__ pre, int N, int relu,
    float* __restrict__ sum, float* __restrict__ sq)
{
    int f  = threadIdx.x;
    int r0 = blockIdx.x * 256;
    int r1 = r0 + 256; if (r1 > N) r1 = N;
    float s = 0.0f, q = 0.0f;
    for (int r = r0; r < r1; ++r) {
        float v = pre[(long long)r * 128 + f];
        if (relu) v = fmaxf(v, 0.0f);
        s += v;
        q += v * v;
    }
    atomicAdd(&sum[f], s);
    atomicAdd(&sq[f], q);
}

// ---------------------------------------------------------------------------
// BN apply + concat: h[n, 0:128] = BN(relu(alpha)), h[n,128:256] = BN(beta)
// stats layout: [sum_a(128) | sq_a(128) | sum_b(128) | sq_b(128)]
// bn layout:    [gamma_a(128) | beta_a(128) | gamma_b(128) | beta_b(128)]
// ---------------------------------------------------------------------------
__global__ void bn_apply(const float* __restrict__ apre,
                         const float* __restrict__ bpre,
                         const float* __restrict__ stats,
                         const float* __restrict__ bn,
                         float* __restrict__ h, int N)
{
    long long idx = (long long)blockIdx.x * blockDim.x + threadIdx.x;
    if (idx >= (long long)N * 256) return;
    int n = (int)(idx >> 8);
    int c = (int)(idx & 255);
    float invN = 1.0f / (float)N;
    float v, mu, var, g, b;
    if (c < 128) {
        int f = c;
        v   = fmaxf(apre[(long long)n * 128 + f], 0.0f);
        mu  = stats[f] * invN;
        var = stats[128 + f] * invN - mu * mu;
        g   = bn[f];
        b   = bn[128 + f];
    } else {
        int f = c - 128;
        v   = bpre[(long long)n * 128 + f];
        mu  = stats[256 + f] * invN;
        var = stats[384 + f] * invN - mu * mu;
        g   = bn[256 + f];
        b   = bn[384 + f];
    }
    h[idx] = g * (v - mu) * rsqrtf(var + 1e-3f) + b;
}

// ---------------------------------------------------------------------------
// z = h(N x 256) @ We(256 x 16) + be
// ---------------------------------------------------------------------------
__global__ void dense_z(const float* __restrict__ h,
                        const float* __restrict__ We,
                        const float* __restrict__ be,
                        float* __restrict__ z, int N)
{
    long long idx = (long long)blockIdx.x * blockDim.x + threadIdx.x;
    if (idx >= (long long)N * 16) return;
    int n = (int)(idx >> 4);
    int m = (int)(idx & 15);
    float s = be[m];
    const float* hr = h + (long long)n * 256;
#pragma unroll 8
    for (int k = 0; k < 256; ++k) s += hr[k] * We[k * 16 + m];
    z[idx] = s;
}

// ---------------------------------------------------------------------------
// Per-edge softmax over M=16: s = val * (z[row] + z[col]); out = softmax(s)
// ---------------------------------------------------------------------------
__global__ void edge_softmax(const int* __restrict__ rows,
                             const int* __restrict__ cols,
                             const float* __restrict__ vals,
                             const float* __restrict__ z,
                             float* __restrict__ out, int E)
{
    int e = blockIdx.x * blockDim.x + threadIdx.x;
    if (e >= E) return;
    const float* zr = z + (long long)rows[e] * 16;
    const float* zc = z + (long long)cols[e] * 16;
    float v = vals[e];
    float s[16];
    float mx = -3.4e38f;
#pragma unroll
    for (int m = 0; m < 16; ++m) {
        s[m] = v * (zr[m] + zc[m]);
        mx = fmaxf(mx, s[m]);
    }
    float tot = 0.0f;
#pragma unroll
    for (int m = 0; m < 16; ++m) {
        s[m] = __expf(s[m] - mx);
        tot += s[m];
    }
    float r = 1.0f / tot;
    float* o = out + (long long)e * 16;
#pragma unroll
    for (int m = 0; m < 16; ++m) o[m] = s[m] * r;
}

// ---------------------------------------------------------------------------
// Host-side orchestration
// ---------------------------------------------------------------------------
static inline void launch_zero(float* p, long long n, hipStream_t st) {
    int blocks = (int)((n + 255) / 256);
    zero_f32<<<blocks, 256, 0, st>>>(p, n);
}

extern "C" void kernel_launch(void* const* d_in, const int* in_sizes, int n_in,
                              void* d_out, int out_size, void* d_ws, size_t ws_size,
                              hipStream_t stream) {
    const float* x    = (const float*)d_in[0];
    const int*   rows = (const int*)d_in[1];
    const int*   cols = (const int*)d_in[2];
    const float* vals = (const float*)d_in[3];
    const float* l0Wa = (const float*)d_in[4];
    const float* l0Wb = (const float*)d_in[5];
    const float* l0bn = (const float*)d_in[6];
    const float* l1Wa = (const float*)d_in[7];
    const float* l1Wb = (const float*)d_in[8];
    const float* l1bn = (const float*)d_in[9];
    const float* We   = (const float*)d_in[10];
    const float* be   = (const float*)d_in[11];
    float* out = (float*)d_out;

    const int N = in_sizes[0] / 128;   // 50000
    const int E = in_sizes[1];         // 800000

    // ---- workspace carve-up ----
    const long long nf256 = (long long)N * 256;
    char* ws = (char*)d_ws;
    size_t off = 0;
    auto carve = [&](size_t bytes) {
        void* p = ws + off;
        off += (bytes + 255) & ~(size_t)255;
        return p;
    };
    float* h   = (float*)carve(nf256 * 4);
    float* a0  = (float*)carve(nf256 * 4);
    float* a1  = (float*)carve(nf256 * 4);
    float* a2  = (float*)carve(nf256 * 4);
    float* tmp = (float*)carve(nf256 * 4);      // also alpha_pre/beta_pre
    float* zbuf  = (float*)carve((long long)N * 16 * 4);
    float* stats = (float*)carve(512 * 4);
    unsigned short* Wsa = (unsigned short*)carve((size_t)4 * 256 * 128 * 2);
    unsigned short* Wsb = (unsigned short*)carve((size_t)4 * 256 * 128 * 2);

    float* alpha_pre = tmp;                          // N x 128
    float* beta_pre  = tmp + (long long)N * 128;     // N x 128

    const int spmmBlocks = (E * 32 + 255) / 256;     // one wave per edge
    const int gemmBlocks = (N + 127) / 128;
    const int statBlocks = (N + 255) / 256;
    const long long applyN = (long long)N * 256;

    // ================= layer 0 (Kin = 128) and layer 1 (Kin = 256) ==========
    for (int layer = 0; layer < 2; ++layer) {
        const int Kin = (layer == 0) ? 128 : 256;
        const int kshift = (layer == 0) ? 2 : 3;     // log2(Kin/32)
        const int kmask  = (1 << kshift) - 1;
        const float* xin = (layer == 0) ? x : h;
        const float* Wa  = (layer == 0) ? l0Wa : l1Wa;
        const float* Wb  = (layer == 0) ? l0Wb : l1Wb;
        const float* bn  = (layer == 0) ? l0bn : l1bn;
        const long long nk = (long long)N * Kin;

        // a0 = A x ; a1 = A a0 ; a2 = A (A a1)
        launch_zero(a0, nk, stream);
        spmm_atomic<<<spmmBlocks, 256, 0, stream>>>(rows, cols, vals, xin, a0, E, Kin);
        launch_zero(a1, nk, stream);
        spmm_atomic<<<spmmBlocks, 256, 0, stream>>>(rows, cols, vals, a0, a1, E, Kin);
        launch_zero(tmp, nk, stream);
        spmm_atomic<<<spmmBlocks, 256, 0, stream>>>(rows, cols, vals, a1, tmp, E, Kin);
        launch_zero(a2, nk, stream);
        spmm_atomic<<<spmmBlocks, 256, 0, stream>>>(rows, cols, vals, tmp, a2, E, Kin);

        // pack weights to bf16 transposed-block stacks
        {
            int total = 4 * Kin * 128;
            int wb = (total + 255) / 256;
            wprep<<<wb, 256, 0, stream>>>(Wa, Wsa, Kin);
            wprep<<<wb, 256, 0, stream>>>(Wb, Wsb, Kin);
        }

        // alpha_pre / beta_pre via WMMA GEMM (tmp region is free again here)
        gemm_wmma_bf16<<<gemmBlocks, 256, 0, stream>>>(
            xin, a0, a1, a2, Wsa, alpha_pre, N, Kin, kshift, kmask);
        gemm_wmma_bf16<<<gemmBlocks, 256, 0, stream>>>(
            xin, a0, a1, a2, Wsb, beta_pre, N, Kin, kshift, kmask);

        // batch norm (training-mode stats) + concat into h
        launch_zero(stats, 512, stream);
        bn_stats<<<statBlocks, 128, 0, stream>>>(alpha_pre, N, 1, stats,       stats + 128);
        bn_stats<<<statBlocks, 128, 0, stream>>>(beta_pre,  N, 0, stats + 256, stats + 384);
        bn_apply<<<(int)((applyN + 255) / 256), 256, 0, stream>>>(
            alpha_pre, beta_pre, stats, bn, h, N);
    }

    // ================= edge dense + softmax ================================
    dense_z<<<(int)(((long long)N * 16 + 255) / 256), 256, 0, stream>>>(h, We, be, zbuf, N);
    edge_softmax<<<(E + 255) / 256, 256, 0, stream>>>(rows, cols, vals, zbuf, out, E);

    (void)n_in; (void)out_size; (void)ws_size;
}
